// MarlinRazerLinear_69269232550409
// MI455X (gfx1250) — matrix operand order repaired and checked by reading, not compiled, not run on hardware
//
#include <hip/hip_runtime.h>

typedef __attribute__((ext_vector_type(16))) _Float16     v16h;
typedef __attribute__((ext_vector_type(8)))  _Float16     v8h;
typedef __attribute__((ext_vector_type(8)))  float        v8f;
typedef __attribute__((ext_vector_type(4)))  unsigned int v4u;
typedef __attribute__((ext_vector_type(8)))  unsigned int v8u;

#define KDIM   4096
#define NDIM   11008
#define BLK_M  128
#define BLK_N  128
#define BLK_K  64
#define NCHUNK (KDIM / BLK_K)

// GEMM runs in a permuted K-space: within every aligned 8-K group, position j
// holds original K = pi(j), pi = [0,4,1,5,2,6,3,7]. B gets this order for free
// from nibble mask-extraction; A is permuted once while staging into LDS.
// Group scales (128-wide) are unaffected by the permutation.

__global__ __launch_bounds__(256) void marlin_razer_wmma(
    const _Float16* __restrict__ xg,        // [M, K] fp16
    const unsigned int* __restrict__ bq,    // [K/8, N] packed int4
    const _Float16* __restrict__ sg,        // [K/128, N] fp16 scales
    _Float16* __restrict__ yg)              // [M, N] fp16
{
    __shared__ __align__(16) _Float16     As[2][BLK_M][BLK_K];     // 32 KB
    __shared__ __align__(16) unsigned int Bw[2][BLK_K / 8][BLK_N]; //  8 KB

    const int tid    = threadIdx.x;
    const int lane   = tid & 31;
    const int wave   = tid >> 5;    // 0..7 -> 16 N-columns each
    const int lane15 = lane & 15;
    const int laneHi = lane >> 4;   // 0 or 1

    const int n0 = blockIdx.x * BLK_N;
    const int m0 = blockIdx.y * BLK_M;

    auto load_tiles = [&](int buf, int kc) {
        // A tile: 128 rows x 64 halves; permute halves [0,4,1,5,2,6,3,7] per 8-group
        #pragma unroll
        for (int i = 0; i < 4; ++i) {
            int idx  = tid + i * 256;
            int row  = idx >> 3;
            int col8 = (idx & 7) << 3;
            v4u raw  = *(const v4u*)(xg + (size_t)(m0 + row) * KDIM + kc + col8);
            v8h h    = __builtin_bit_cast(v8h, raw);
            v8h perm = __builtin_shufflevector(h, h, 0, 4, 1, 5, 2, 6, 3, 7);
            *(v4u*)(&As[buf][row][col8]) = __builtin_bit_cast(v4u, perm);
        }
        // packed-B tile: 8 word-rows x 128 cols; async DMA memory->LDS, no VGPR data
        {
            int r  = tid >> 5;
            int nc = (tid & 31) << 2;
            const unsigned int* gsrc = bq + (size_t)((kc >> 3) + r) * NDIM + n0 + nc;
            unsigned int ldsaddr = (unsigned int)(size_t)(&Bw[buf][r][nc]);
            asm volatile("global_load_async_to_lds_b128 %0, %1, off"
                         :: "v"(ldsaddr), "v"(gsrc)
                         : "memory");
        }
    };

    const v8f vzero = {0.f, 0.f, 0.f, 0.f, 0.f, 0.f, 0.f, 0.f};
    v8f acc[8];
    #pragma unroll
    for (int mt = 0; mt < 8; ++mt) acc[mt] = vzero;

    load_tiles(0, 0);

    const size_t scol = (size_t)(n0 + wave * 16 + lane15);
    _Float16 sc_cur = sg[scol];  // group 0 scale

    for (int c = 0; c < NCHUNK; ++c) {
        const int kc  = c * BLK_K;
        const int buf = c & 1;

        // our async B loads into tile[buf] must have landed before anyone reads
        asm volatile("s_wait_asynccnt 0x0" ::: "memory");
        __syncthreads();

        if (c + 1 < NCHUNK) {
            load_tiles(buf ^ 1, kc + BLK_K);
            if (c + 2 < NCHUNK) {
                // warm L2 two chunks ahead (global_prefetch_b8)
                __builtin_prefetch(xg + (size_t)(m0 + (tid >> 1)) * KDIM + kc + 2 * BLK_K, 0, 1);
                __builtin_prefetch(bq + (size_t)(((kc + 2 * BLK_K) >> 3) + (tid >> 5)) * NDIM
                                      + n0 + ((tid & 31) << 2), 0, 1);
            }
        }

        // software-pipelined group scale: issue next chunk's load now, use current
        _Float16 sc_nxt = sc_cur;
        if (c + 1 < NCHUNK)
            sc_nxt = sg[(size_t)((kc + BLK_K) >> 7) * NDIM + scol];

        const _Float16 sc  = sc_cur;
        const _Float16 off = sc * (_Float16)(-1032.0f);  // folds the "-8" zero point
        v16h mscV, maddV;
        #pragma unroll
        for (int j = 0; j < 16; ++j) { mscV[j] = sc; maddV[j] = off; }

        #pragma unroll
        for (int ks = 0; ks < BLK_K; ks += 32) {
            // ---- B fragment: 2 packed words -> 16 fp16 (bit-trick, packed fma) ----
            const int nn = wave * 16 + lane15;
            const int rw = (ks >> 3) + (laneHi << 1);     // K base = ks + laneHi*16
            const unsigned int w0 = Bw[buf][rw][nn];      // K' = base .. base+7
            const unsigned int w1 = Bw[buf][rw + 1][nn];  // K' = base+8 .. base+15
            v8u braw;
            braw[0] = ( w0        & 0x000F000Fu) | 0x64006400u;  // fp16: 1024+q
            braw[1] = ((w0 >>  4) & 0x000F000Fu) | 0x64006400u;
            braw[2] = ((w0 >>  8) & 0x000F000Fu) | 0x64006400u;
            braw[3] = ((w0 >> 12) & 0x000F000Fu) | 0x64006400u;
            braw[4] = ( w1        & 0x000F000Fu) | 0x64006400u;
            braw[5] = ((w1 >>  4) & 0x000F000Fu) | 0x64006400u;
            braw[6] = ((w1 >>  8) & 0x000F000Fu) | 0x64006400u;
            braw[7] = ((w1 >> 12) & 0x000F000Fu) | 0x64006400u;
            // (h - 1032) * sc  ==  h*sc + (-1032*sc)   -> 8x v_pk_fma_f16
            const v16h b = __builtin_elementwise_fma(
                __builtin_bit_cast(v16h, braw), mscV, maddV);

            // ---- A fragments + WMMA: 8 M-tiles against this one B fragment ----
            const int kb = ks + (laneHi << 4);
            #pragma unroll
            for (int mt = 0; mt < 8; ++mt) {
                const int m = mt * 16 + lane15;
                v8h lo = *(const v8h*)(&As[buf][m][kb]);
                v8h hi = *(const v8h*)(&As[buf][m][kb + 8]);
                v16h a = __builtin_shufflevector(lo, hi,
                          0, 1, 2, 3, 4, 5, 6, 7, 8, 9, 10, 11, 12, 13, 14, 15);
                acc[mt] = __builtin_amdgcn_wmma_f32_16x16x32_f16(
                    false, a, false, b, (short)0, acc[mt], false, false);
            }
        }

        sc_cur = sc_nxt;
    }

    // Epilogue: C layout — VGPR r: lanes0-15 (M=r, N=lane), lanes16-31 (M=r+8, N=lane-16)
    #pragma unroll
    for (int mt = 0; mt < 8; ++mt) {
        const int n  = n0 + wave * 16 + lane15;
        const int mb = m0 + mt * 16 + laneHi * 8;
        #pragma unroll
        for (int r = 0; r < 8; ++r)
            yg[(size_t)(mb + r) * NDIM + n] = (_Float16)acc[mt][r];
    }
}

extern "C" void kernel_launch(void* const* d_in, const int* in_sizes, int n_in,
                              void* d_out, int out_size, void* d_ws, size_t ws_size,
                              hipStream_t stream) {
    (void)n_in; (void)out_size; (void)d_ws; (void)ws_size;
    const _Float16*     x  = (const _Float16*)d_in[0];
    const unsigned int* bq = (const unsigned int*)d_in[1];
    const _Float16*     s  = (const _Float16*)d_in[2];
    _Float16*           y  = (_Float16*)d_out;

    const int M_total = in_sizes[0] / KDIM;        // 8*512 = 4096
    dim3 grid(NDIM / BLK_N, M_total / BLK_M);      // 86 x 32
    dim3 block(256);                               // 8 waves (wave32)
    marlin_razer_wmma<<<grid, block, 0, stream>>>(x, bq, s, y);
}
